// EncoderLayer_87342454931930
// MI455X (gfx1250) — matrix operand order repaired and checked by reading, not compiled, not run on hardware
//
#include <hip/hip_runtime.h>
#include <hip/hip_fp16.h>
#include <math.h>

#define HIDDEN 256
#define HEADS 8
#define DHEAD 32

typedef _Float16 v16h __attribute__((ext_vector_type(16)));
typedef _Float16 v8h  __attribute__((ext_vector_type(8)));
typedef float    v8f  __attribute__((ext_vector_type(8)));

union ABFrag { v16h v; v8h h[2]; };

// ---------------------------------------------------------------------------
// Weight convert: f32 [k][n] -> f16 transposed [n][k] (256x256)
// ---------------------------------------------------------------------------
__global__ void convert_wt(const float* __restrict__ W, _Float16* __restrict__ Wt) {
    int idx = blockIdx.x * blockDim.x + threadIdx.x;   // 65536 threads
    int k = idx >> 8, n = idx & 255;
    Wt[n * HIDDEN + k] = (_Float16)W[idx];
}

// ---------------------------------------------------------------------------
// LayerNorm: one wave per row (wave32), 8 elems/lane, shuffle reduction -> f16
// ---------------------------------------------------------------------------
__global__ void ln_kernel(const float* __restrict__ x, const float* __restrict__ w,
                          const float* __restrict__ b, _Float16* __restrict__ y, int nrows) {
    int row  = blockIdx.x * 8 + (threadIdx.x >> 5);
    int lane = threadIdx.x & 31;
    if (row >= nrows) return;
    const float* xr = x + (size_t)row * HIDDEN;
    float vals[8];
    float s = 0.f;
#pragma unroll
    for (int i = 0; i < 8; ++i) { vals[i] = xr[lane + 32 * i]; s += vals[i]; }
#pragma unroll
    for (int m = 16; m; m >>= 1) s += __shfl_xor(s, m, 32);
    float mean = s * (1.f / HIDDEN);
    float vsum = 0.f;
#pragma unroll
    for (int i = 0; i < 8; ++i) { float d = vals[i] - mean; vsum += d * d; }
#pragma unroll
    for (int m = 16; m; m >>= 1) vsum += __shfl_xor(vsum, m, 32);
    float rstd = rsqrtf(vsum * (1.f / HIDDEN) + 1e-5f);
    _Float16* yr = y + (size_t)row * HIDDEN;
#pragma unroll
    for (int i = 0; i < 8; ++i) {
        int c = lane + 32 * i;
        yr[c] = (_Float16)((vals[i] - mean) * rstd * w[c] + b[c]);
    }
}

// ---------------------------------------------------------------------------
// WMMA GEMM: C[M x 256] = A[M x 256](f16) @ W(256x256, f16 stored transposed
// as Bt[n][k]) + bias, with fused epilogue.
// Block = 256 threads = 8 waves; wave computes 16(M) x 64(N); block covers
// 32(M) x 256(N). K loop: 8 steps of v_wmma_f32_16x16x32_f16, A frag reused
// across the wave's 4 N-tiles.
//   EPI 0: store f16            EPI 1: outF = res + val (f32)
//   EPI 2: exact GELU -> f16
// ---------------------------------------------------------------------------
template <int EPI>
__global__ __launch_bounds__(256) void gemm256(
        const _Float16* __restrict__ A, const _Float16* __restrict__ Bt,
        const float* __restrict__ bias, int M,
        _Float16* __restrict__ outH, float* __restrict__ outF,
        const float* __restrict__ res) {
    int tid  = threadIdx.x;
    int wave = tid >> 5, lane = tid & 31;
    int mbase = blockIdx.x * 32 + (wave & 1) * 16;
    int nbase = (wave >> 1) * 64;
    if (mbase >= M) return;
    int lh = lane >> 4;   // 0 or 1 (half-wave)
    int ll = lane & 15;

    v8f acc[4];
#pragma unroll
    for (int nt = 0; nt < 4; ++nt) acc[nt] = (v8f){};

    const _Float16* arow = A + (size_t)(mbase + ll) * HIDDEN;
#pragma unroll
    for (int kt = 0; kt < 8; ++kt) {
        // A frag 16x32 f16: lane ll -> M row; elems 0..7 : K = kt*32+lh*8+j,
        // elems 8..15 : K = kt*32+16+lh*8+j
        int ka = kt * 32 + lh * 8;
        ABFrag af;
        af.h[0] = *(const v8h*)(arow + ka);
        af.h[1] = *(const v8h*)(arow + ka + 16);
        // B frag 32x16 f16: lane ll -> N col; same K mapping as A frag
        int kb = kt * 32 + lh * 8;
#pragma unroll
        for (int nt = 0; nt < 4; ++nt) {
            const _Float16* brow = Bt + (size_t)(nbase + nt * 16 + ll) * HIDDEN + kb;
            ABFrag bf;
            bf.h[0] = *(const v8h*)(brow);
            bf.h[1] = *(const v8h*)(brow + 16);
            acc[nt] = __builtin_amdgcn_wmma_f32_16x16x32_f16(
                false, af.v, false, bf.v, (short)0, acc[nt], false, false);
        }
    }
    // C/D layout: lane, VGPR i -> M = mbase + lh*8 + i, N = nbase + nt*16 + ll
#pragma unroll
    for (int nt = 0; nt < 4; ++nt) {
        int n = nbase + nt * 16 + ll;
        float bn = bias[n];
#pragma unroll
        for (int i = 0; i < 8; ++i) {
            int m = mbase + lh * 8 + i;
            float val = acc[nt][i] + bn;
            size_t o = (size_t)m * HIDDEN + n;
            if (EPI == 0) {
                outH[o] = (_Float16)val;
            } else if (EPI == 1) {
                outF[o] = res[o] + val;
            } else {  // exact GELU
                val = 0.5f * val * (1.f + erff(val * 0.70710678118f));
                outH[o] = (_Float16)val;
            }
        }
    }
}

// ---------------------------------------------------------------------------
// Segment-softmax init: acc=0 over NR*256; z=0, smax=-inf(enc) over NR*8
// ---------------------------------------------------------------------------
__global__ void seg_init(float* __restrict__ acc, float* __restrict__ z,
                         int* __restrict__ smax, int nrow) {
    int idx = blockIdx.x * blockDim.x + threadIdx.x;
    if (idx < nrow * HIDDEN) acc[idx] = 0.f;
    if (idx < nrow * HEADS) { z[idx] = 0.f; smax[idx] = (int)0x80000000; }
}

__device__ __forceinline__ int enc_f32(float f) {
    int i = __float_as_int(f);
    return i >= 0 ? i : (i ^ 0x7FFFFFFF);   // order-preserving int encoding
}
__device__ __forceinline__ float dec_f32(int i) {
    return __int_as_float(i >= 0 ? i : (i ^ 0x7FFFFFFF));
}

// ---------------------------------------------------------------------------
// Edge pass 1: s[e,h] = (q[row]·k[col]) * scale + bias[e,h]; segment max
// thread per (edge, head); q/k rows are 64B-aligned f16
// ---------------------------------------------------------------------------
__global__ void edge_pass1(const _Float16* __restrict__ q, const _Float16* __restrict__ k,
                           const float* __restrict__ bias, const int* __restrict__ row,
                           const int* __restrict__ col, float* __restrict__ sbuf,
                           int* __restrict__ smax, int nedge) {
    int idx = blockIdx.x * blockDim.x + threadIdx.x;
    if (idx >= nedge * HEADS) return;
    int e = idx >> 3, h = idx & 7;
    int r = row[e], c = col[e];
    const _Float16* qp = q + (size_t)r * HIDDEN + h * DHEAD;
    const _Float16* kp = k + (size_t)c * HIDDEN + h * DHEAD;
    float s = 0.f;
#pragma unroll
    for (int t = 0; t < 4; ++t) {
        v8h qa = *(const v8h*)(qp + t * 8);
        v8h kb = *(const v8h*)(kp + t * 8);
#pragma unroll
        for (int j = 0; j < 8; ++j) s += (float)qa[j] * (float)kb[j];
    }
    s = s * 0.17677669529663687f + bias[idx];   // DHEAD^-0.5
    sbuf[idx] = s;
    atomicMax(&smax[r * HEADS + h], enc_f32(s));
}

// ---------------------------------------------------------------------------
// Edge pass 2: e = exp(s - smax); z[r,h] += e; acc[r,:] += e * v[c,:]
// thread per (edge, dim) so v-gathers and acc-scatters are coalesced
// ---------------------------------------------------------------------------
__global__ void edge_pass2(const float* __restrict__ sbuf, const int* __restrict__ smax,
                           const int* __restrict__ row, const int* __restrict__ col,
                           const _Float16* __restrict__ v, float* __restrict__ z,
                           float* __restrict__ acc, int nedge) {
    int idx = blockIdx.x * 256 + threadIdx.x;   // nedge*256 total
    int e = idx >> 8, d = idx & 255;
    if (e >= nedge) return;
    int h = d >> 5;
    int r = row[e], c = col[e];
    float m  = dec_f32(smax[r * HEADS + h]);
    float ex = __expf(sbuf[e * HEADS + h] - m);
    if ((d & 31) == 0) atomicAdd(&z[r * HEADS + h], ex);
    atomicAdd(&acc[(size_t)r * HIDDEN + d], ex * (float)v[(size_t)c * HIDDEN + d]);
}

// ---------------------------------------------------------------------------
// Normalize: attn[r,d] = acc[r,d] / (z[r,h] + eps) -> f16 (input to O GEMM)
// ---------------------------------------------------------------------------
__global__ void attn_norm(const float* __restrict__ acc, const float* __restrict__ z,
                          _Float16* __restrict__ out, int total) {
    int idx = blockIdx.x * blockDim.x + threadIdx.x;
    if (idx >= total) return;
    int r = idx >> 8, h = (idx >> 5) & 7;
    out[idx] = (_Float16)(acc[idx] / (z[r * HEADS + h] + 1e-16f));
}

// ---------------------------------------------------------------------------
extern "C" void kernel_launch(void* const* d_in, const int* in_sizes, int n_in,
                              void* d_out, int out_size, void* d_ws, size_t ws_size,
                              hipStream_t stream) {
    (void)n_in; (void)out_size; (void)ws_size;
    const float* x_all = (const float*)d_in[0];
    const float* biasT = (const float*)d_in[1];
    const float* biasC = (const float*)d_in[2];
    const int*   trow  = (const int*)d_in[3];
    const int*   tcol  = (const int*)d_in[4];
    const int*   crow  = (const int*)d_in[5];
    const int*   ccol  = (const int*)d_in[6];
    const float* ln1w = (const float*)d_in[9];
    const float* ln1b = (const float*)d_in[10];
    const float* Wq1 = (const float*)d_in[11]; const float* bq1 = (const float*)d_in[12];
    const float* Wk1 = (const float*)d_in[13]; const float* bk1 = (const float*)d_in[14];
    const float* Wv1 = (const float*)d_in[15]; const float* bv1 = (const float*)d_in[16];
    const float* Wo1 = (const float*)d_in[17]; const float* bo1 = (const float*)d_in[18];
    const float* Wq2 = (const float*)d_in[19]; const float* bq2 = (const float*)d_in[20];
    const float* Wk2 = (const float*)d_in[21]; const float* bk2 = (const float*)d_in[22];
    const float* Wv2 = (const float*)d_in[23]; const float* bv2 = (const float*)d_in[24];
    const float* Wo2 = (const float*)d_in[25]; const float* bo2 = (const float*)d_in[26];
    const float* ln2w = (const float*)d_in[27];
    const float* ln2b = (const float*)d_in[28];
    const float* Wf1 = (const float*)d_in[29]; const float* bf1 = (const float*)d_in[30];
    const float* Wf2 = (const float*)d_in[31]; const float* bf2 = (const float*)d_in[32];

    const int ET   = in_sizes[3];
    const int EC   = in_sizes[5];
    const int NALL = in_sizes[0] / HIDDEN;
    const int NT   = NALL / 2;
    const int NC   = NALL - NT;
    const int NR   = (NT > NC) ? NT : NC;

    // ---- workspace layout (bytes) ----
    char* ws = (char*)d_ws;
    const size_t WSZ = (size_t)65536 * 2;           // one f16 256x256 weight
    _Float16* wt = (_Float16*)ws;                   // 10 matrices
    size_t off = 10 * WSZ;
    _Float16* yH = (_Float16*)(ws + off); off += (size_t)NALL * HIDDEN * 2;   // LN out (reused for LN2)
    _Float16* q1H = (_Float16*)(ws + off); off += (size_t)NT * HIDDEN * 2;
    _Float16* k1H = (_Float16*)(ws + off); off += (size_t)NC * HIDDEN * 2;
    _Float16* v1H = (_Float16*)(ws + off); off += (size_t)NC * HIDDEN * 2;
    _Float16* q2H = (_Float16*)(ws + off); off += (size_t)NC * HIDDEN * 2;
    _Float16* k2H = (_Float16*)(ws + off); off += (size_t)NT * HIDDEN * 2;
    _Float16* v2H = (_Float16*)(ws + off); off += (size_t)NT * HIDDEN * 2;
    float* sbufF  = (float*)(ws + off);
    _Float16* attnH = (_Float16*)(ws + off);        // aliases sbuf (dead by then)
    { size_t se = (size_t)((ET > EC) ? ET : EC) * HEADS * 4;
      size_t ae = (size_t)NR * HIDDEN * 2;
      off += (se > ae) ? se : ae; }
    int*   smaxI = (int*)(ws + off);  off += (size_t)NR * HEADS * 4;
    float* zF    = (float*)(ws + off); off += (size_t)NR * HEADS * 4;
    float* accF  = (float*)(ws + off); off += (size_t)NR * HIDDEN * 4;
    float* xF    = (float*)(ws + off); off += (size_t)NALL * HIDDEN * 4;
    _Float16* hH = q1H;                             // FFN hidden reuses dead qkv region

    const _Float16* wtQ1 = wt + 0 * 65536; const _Float16* wtK1 = wt + 1 * 65536;
    const _Float16* wtV1 = wt + 2 * 65536; const _Float16* wtO1 = wt + 3 * 65536;
    const _Float16* wtQ2 = wt + 4 * 65536; const _Float16* wtK2 = wt + 5 * 65536;
    const _Float16* wtV2 = wt + 6 * 65536; const _Float16* wtO2 = wt + 7 * 65536;
    const _Float16* wtF1 = wt + 8 * 65536; const _Float16* wtF2 = wt + 9 * 65536;

    // 1) weight conversion (f32 -> f16, transposed)
    const float* Ws[10] = {Wq1, Wk1, Wv1, Wo1, Wq2, Wk2, Wv2, Wo2, Wf1, Wf2};
    for (int i = 0; i < 10; ++i)
        convert_wt<<<256, 256, 0, stream>>>(Ws[i], wt + (size_t)i * 65536);

    // 2) LN1 over all rows -> f16
    ln_kernel<<<(NALL + 7) / 8, 256, 0, stream>>>(x_all, ln1w, ln1b, yH, NALL);

    // 3) QKV GEMMs (WMMA), f16 outputs
    const _Float16* yTgt = yH;
    const _Float16* yCtx = yH + (size_t)NT * HIDDEN;
    gemm256<0><<<(NT + 31) / 32, 256, 0, stream>>>(yTgt, wtQ1, bq1, NT, q1H, nullptr, nullptr);
    gemm256<0><<<(NC + 31) / 32, 256, 0, stream>>>(yCtx, wtK1, bk1, NC, k1H, nullptr, nullptr);
    gemm256<0><<<(NC + 31) / 32, 256, 0, stream>>>(yCtx, wtV1, bv1, NC, v1H, nullptr, nullptr);
    gemm256<0><<<(NC + 31) / 32, 256, 0, stream>>>(yCtx, wtQ2, bq2, NC, q2H, nullptr, nullptr);
    gemm256<0><<<(NT + 31) / 32, 256, 0, stream>>>(yTgt, wtK2, bk2, NT, k2H, nullptr, nullptr);
    gemm256<0><<<(NT + 31) / 32, 256, 0, stream>>>(yTgt, wtV2, bv2, NT, v2H, nullptr, nullptr);

    // 4) sparse MHA: pass1 (scores+max), pass2 (exp+segsum), norm, O-proj+residual
    auto run_mha = [&](const _Float16* q, const _Float16* k, const _Float16* v,
                       const float* ebias, const int* erow, const int* ecol,
                       int nE, int nRow, const _Float16* wo, const float* bo, int xoff) {
        seg_init<<<nRow, 256, 0, stream>>>(accF, zF, smaxI, nRow);
        edge_pass1<<<(nE * HEADS + 255) / 256, 256, 0, stream>>>(q, k, ebias, erow, ecol,
                                                                 sbufF, smaxI, nE);
        edge_pass2<<<nE, 256, 0, stream>>>(sbufF, smaxI, erow, ecol, v, zF, accF, nE);
        attn_norm<<<nRow, 256, 0, stream>>>(accF, zF, attnH, nRow * HIDDEN);
        gemm256<1><<<(nRow + 31) / 32, 256, 0, stream>>>(
            attnH, wo, bo, nRow, nullptr,
            xF + (size_t)xoff * HIDDEN, x_all + (size_t)xoff * HIDDEN);
    };
    // concat order in reference: rows [0,NC) <- MHA2 output, rows [NC,NALL) <- MHA1 output
    run_mha(q1H, k1H, v1H, biasT, trow, tcol, ET, NT, wtO1, bo1, NC);
    run_mha(q2H, k2H, v2H, biasC, crow, ccol, EC, NC, wtO2, bo2, 0);

    // 5) LN2 on x -> f16 (reuse yH)
    ln_kernel<<<(NALL + 7) / 8, 256, 0, stream>>>(xF, ln2w, ln2b, yH, NALL);

    // 6) FFN: GELU(y @ f1 + b1) @ f2 + b2, residual with x -> d_out (f32)
    gemm256<2><<<(NALL + 31) / 32, 256, 0, stream>>>(yH, wtF1, bf1, NALL, hH, nullptr, nullptr);
    gemm256<1><<<(NALL + 31) / 32, 256, 0, stream>>>(hH, wtF2, bf2, NALL, nullptr,
                                                     (float*)d_out, xF);
}
